// PromptSaturationDetectorV0_43937515438344
// MI455X (gfx1250) — matrix operand order repaired and checked by reading, not compiled, not run on hardware
//
#include <hip/hip_runtime.h>
#include <math.h>

// ---- problem constants (match reference) ----
constexpr int kB = 16;      // batch
constexpr int kT = 2048;    // timesteps
constexpr int kH = 256;     // hidden
constexpr int kF = 1024;    // fan width
constexpr int kM = kB * kT; // 32768 rows for batch GEMMs

typedef __attribute__((ext_vector_type(16))) __bf16 v16bf;
typedef __attribute__((ext_vector_type(8)))  __bf16 v8bf;
typedef __attribute__((ext_vector_type(8)))  float  v8f;

// ---- bf16 <-> f32 bit helpers (RNE) ----
__device__ __forceinline__ unsigned bfbits(float f) {
  unsigned u = __builtin_bit_cast(unsigned, f);
  return ((u + 0x7FFFu + ((u >> 16) & 1u)) >> 16) & 0xFFFFu;
}
__device__ __forceinline__ __bf16 f2bf(float f) {
  return __builtin_bit_cast(__bf16, (unsigned short)bfbits(f));
}
__device__ __forceinline__ unsigned pack2(float lo, float hi) {
  return (bfbits(hi) << 16) | bfbits(lo);
}
__device__ __forceinline__ float bf2f(__bf16 h) {
  unsigned u = ((unsigned)__builtin_bit_cast(unsigned short, h)) << 16;
  return __builtin_bit_cast(float, u);
}
__device__ __forceinline__ float sigm(float x) { return 1.0f / (1.0f + __expf(-x)); }

// A fragment 16x32 (MxK) from row-major bf16: two contiguous 16B chunks/lane.
__device__ __forceinline__ v16bf load_a_frag(const __bf16* __restrict__ A,
                                             int lda, int m0, int k0) {
  const int lane = threadIdx.x & 31;
  const int m = lane & 15, half = lane >> 4;
  const __bf16* p = A + (size_t)(m0 + m) * lda + k0 + half * 16;
  v8bf lo = *(const v8bf*)(p);
  v8bf hi = *(const v8bf*)(p + 8);
  return __builtin_shufflevector(lo, hi, 0, 1, 2, 3, 4, 5, 6, 7,
                                 8, 9, 10, 11, 12, 13, 14, 15);
}

// B fragment 32x16 (KxN) from row-major bf16 W[N,K]: one 32B chunk/lane.
__device__ __forceinline__ v16bf load_b_frag(const __bf16* __restrict__ W,
                                             int ldw, int n0, int k0) {
  const int lane = threadIdx.x & 31;
  const int n = lane & 15, half = lane >> 4;
  return *(const v16bf*)(W + (size_t)(n0 + n) * ldw + k0 + half * 16);
}

// ---------------------------------------------------------------------------
// elementwise f32 -> bf16 (packed b128 in / b128 out); n multiple of 2048
// ---------------------------------------------------------------------------
__global__ void __launch_bounds__(256)
cvt_f32_to_bf16(const float* __restrict__ in, unsigned* __restrict__ out) {
  const long i = ((long)blockIdx.x * 256 + threadIdx.x) * 8;
  const float4 a = *(const float4*)(in + i);
  const float4 b = *(const float4*)(in + i + 4);
  uint4 r;
  r.x = pack2(a.x, a.y);
  r.y = pack2(a.z, a.w);
  r.z = pack2(b.x, b.y);
  r.w = pack2(b.z, b.w);
  *(uint4*)(out + i / 2) = r;
}

// ---------------------------------------------------------------------------
// out[M,N] = A[M,K](bf16) @ W[N,K](bf16)^T + bias[N](f32)
// ACT: 0 none, 1 LeakyReLU(0.2).  OUTBF: write bf16 else f32.
// 256 threads = 8 waves; each wave computes a 16x64 strip (4 tiles, A reuse x4).
// ---------------------------------------------------------------------------
template <int K, int ACT, bool OUTBF>
__global__ void __launch_bounds__(256)
gemm_bias_act(const __bf16* __restrict__ A, const __bf16* __restrict__ W,
              const float* __restrict__ bias, void* __restrict__ outp, int N) {
  const int wid     = blockIdx.x * 8 + (threadIdx.x >> 5);
  const int stripsN = N >> 6;
  const int m0 = (wid / stripsN) << 4;
  const int n0 = (wid % stripsN) << 6;
  const int lane = threadIdx.x & 31;
  const int n = lane & 15, half = lane >> 4;

  v8f acc[4] = {{}, {}, {}, {}};
#pragma unroll 4
  for (int k0 = 0; k0 < K; k0 += 32) {
    __builtin_prefetch(A + (size_t)(m0 + n) * K + k0 + 128, 0, 0);
    v16bf a = load_a_frag(A, K, m0, k0);
#pragma unroll
    for (int i = 0; i < 4; ++i) {
      v16bf b = load_b_frag(W, K, n0 + 16 * i, k0);
      acc[i] = __builtin_amdgcn_wmma_f32_16x16x32_bf16(false, a, false, b,
                                                       (short)0, acc[i],
                                                       false, false);
    }
  }
#pragma unroll
  for (int i = 0; i < 4; ++i) {
    const int nn = n0 + 16 * i + n;
    const float bv = bias[nn];
#pragma unroll
    for (int r = 0; r < 8; ++r) {
      float v = acc[i][r] + bv;
      if (ACT == 1) v = (v >= 0.0f) ? v : 0.2f * v;
      const long idx = (long)(m0 + half * 8 + r) * N + nn;
      if (OUTBF) ((__bf16*)outp)[idx] = f2bf(v);
      else       ((float*)outp)[idx]  = v;
    }
  }
}

// ---------------------------------------------------------------------------
// Persistent LSTM recurrence. Grid = 16 blocks x 128 threads (4 waves).
// Block b owns hidden columns [16*b, 16*b+16). Wave g owns gate g's tile and
// keeps its Whh slice resident as 8 bf16 B-fragments (loaded from HBM once).
// h state double-buffered bf16 in global; one grid barrier per step.
// ---------------------------------------------------------------------------
__global__ void __launch_bounds__(128)
lstm_recurrence(const float* __restrict__ xg,     // [B*T, 4H] f32 (bias incl.)
                const __bf16* __restrict__ Whh,   // [4H, H] bf16
                __bf16* __restrict__ hbuf,        // [2][16][256] bf16, zeroed
                __bf16* __restrict__ hseq,        // [B*T, H] bf16 output
                unsigned* __restrict__ ctr) {     // zeroed barrier counter
  const int tid  = threadIdx.x;
  const int gate = tid >> 5;
  const int lane = tid & 31;
  const int n0   = blockIdx.x << 4;
  const int n = lane & 15, half = lane >> 4;

  __shared__ float gates[4][16][16];
  __shared__ float cstate[16][16];

  v16bf bw[8];
#pragma unroll
  for (int kk = 0; kk < 8; ++kk)
    bw[kk] = load_b_frag(Whh, kH, gate * kH + n0, kk * 32);

  for (int i = tid; i < 256; i += 128) (&cstate[0][0])[i] = 0.0f;
  __syncthreads();

  for (int t = 0; t < kT; ++t) {
    const __bf16* hcur  = hbuf + (size_t)(t & 1) * (16 * kH);
    __bf16*       hnext = hbuf + (size_t)((t + 1) & 1) * (16 * kH);

    v8f acc = {};
#pragma unroll
    for (int kk = 0; kk < 8; ++kk) {
      v16bf a = load_a_frag(hcur, kH, 0, kk * 32);
      acc = __builtin_amdgcn_wmma_f32_16x16x32_bf16(false, a, false, bw[kk],
                                                    (short)0, acc, false, false);
    }

#pragma unroll
    for (int r = 0; r < 8; ++r) {
      const int b = half * 8 + r;
      gates[gate][b][n] =
          acc[r] + xg[(size_t)(b * kT + t) * (4 * kH) + gate * kH + n0 + n];
    }
    __syncthreads();

#pragma unroll
    for (int e = 0; e < 2; ++e) {
      const int idx = tid * 2 + e;
      const int b = idx >> 4, j = idx & 15;
      const float iv = sigm(gates[0][b][j]);
      const float fv = sigm(gates[1][b][j]);
      const float gv = tanhf(gates[2][b][j]);
      const float ov = sigm(gates[3][b][j]);
      const float cc = fv * cstate[b][j] + iv * gv;
      cstate[b][j] = cc;
      const __bf16 hv = f2bf(ov * tanhf(cc));
      hnext[b * kH + n0 + j] = hv;
      hseq[(size_t)(b * kT + t) * kH + n0 + j] = hv;
    }

    __threadfence();
    __syncthreads();
    if (tid == 0) {
      __hip_atomic_fetch_add(ctr, 1u, __ATOMIC_RELEASE, __HIP_MEMORY_SCOPE_AGENT);
      const unsigned target = gridDim.x * (unsigned)(t + 1);
      while (__hip_atomic_load(ctr, __ATOMIC_ACQUIRE, __HIP_MEMORY_SCOPE_AGENT) <
             target) { }
    }
    __syncthreads();
  }
}

// ---------------------------------------------------------------------------
// head: out[b] = sigmoid( hseq[b, T-1, :] . head_W + head_b )
// ---------------------------------------------------------------------------
__global__ void __launch_bounds__(256)
head_kernel(const __bf16* __restrict__ seq, const float* __restrict__ hw,
            const float* __restrict__ hb, float* __restrict__ out) {
  __shared__ float red[256];
  const int tid = threadIdx.x;
  const float w = hw[tid];
  for (int b = 0; b < kB; ++b) {
    red[tid] = bf2f(seq[(size_t)(b * kT + (kT - 1)) * kH + tid]) * w;
    __syncthreads();
    for (int s = 128; s > 0; s >>= 1) {
      if (tid < s) red[tid] += red[tid + s];
      __syncthreads();
    }
    if (tid == 0) out[b] = sigm(red[0] + hb[0]);
    __syncthreads();
  }
}

// ---------------------------------------------------------------------------
extern "C" void kernel_launch(void* const* d_in, const int* in_sizes, int n_in,
                              void* d_out, int out_size, void* d_ws,
                              size_t ws_size, hipStream_t stream) {
  (void)in_sizes; (void)n_in; (void)out_size; (void)ws_size;
  const float* x      = (const float*)d_in[0];
  const float* l0_Wih = (const float*)d_in[1];  // [4, 4H, H]
  const float* l0_Whh = (const float*)d_in[2];  // [4, 4H, H]
  const float* l0_b   = (const float*)d_in[3];  // [4, 4H]
  const float* fo_W   = (const float*)d_in[4];  // [F, H]
  const float* fo_b   = (const float*)d_in[5];  // [F]
  const float* fi_W   = (const float*)d_in[6];  // [H, F]
  const float* fi_b   = (const float*)d_in[7];  // [H]
  const float* l1_Wih = (const float*)d_in[8];  // [4H, H]
  const float* l1_Whh = (const float*)d_in[9];  // [4H, H]
  const float* l1_b   = (const float*)d_in[10]; // [4H]
  const float* hd_W   = (const float*)d_in[11]; // [1, H]
  const float* hd_b   = (const float*)d_in[12]; // [1]
  float* out = (float*)d_out;

  // ---- workspace layout (bytes) ----
  constexpr size_t WSZ       = (size_t)(4 * kH) * kH;        // 262144 elems/weight
  constexpr size_t XG_BYTES  = (size_t)kM * (4 * kH) * 4;    // 128 MB f32 (also bf16 "mid")
  constexpr size_t SEQ_BYTES = (size_t)kM * kH * 2;          // 16 MB bf16
  constexpr size_t WB_ELEMS  = 4 * WSZ + 4 * WSZ + WSZ + WSZ + WSZ + WSZ; // 3.1M
  constexpr size_t HBUF_BYTES = 2 * 16 * kH * 2;             // 16 KB bf16

  char* ws = (char*)d_ws;
  float*  xg   = (float*)(ws);
  __bf16* mid  = (__bf16*)(ws);                                  // aliases xg
  __bf16* xbf  = (__bf16*)(ws + XG_BYTES);
  __bf16* seqA = (__bf16*)(ws + XG_BYTES + SEQ_BYTES);
  __bf16* seqB = (__bf16*)(ws + XG_BYTES + 2 * SEQ_BYTES);
  __bf16* wbf  = (__bf16*)(ws + XG_BYTES + 3 * SEQ_BYTES);
  __bf16* hbuf = (__bf16*)(ws + XG_BYTES + 3 * SEQ_BYTES + WB_ELEMS * 2);
  unsigned* ctr =
      (unsigned*)(ws + XG_BYTES + 3 * SEQ_BYTES + WB_ELEMS * 2 + HBUF_BYTES);

  __bf16* w_l0_Wih = wbf;                 // 4 x [4H,H]
  __bf16* w_l0_Whh = w_l0_Wih + 4 * WSZ;  // 4 x [4H,H]
  __bf16* w_fo     = w_l0_Whh + 4 * WSZ;  // [F,H]
  __bf16* w_fi     = w_fo + WSZ;          // [H,F]
  __bf16* w_l1_Wih = w_fi + WSZ;          // [4H,H]
  __bf16* w_l1_Whh = w_l1_Wih + WSZ;      // [4H,H]

  auto cvt = [&](const float* src, __bf16* dst, long nelem) {
    cvt_f32_to_bf16<<<(int)(nelem / 2048), 256, 0, stream>>>(src,
                                                             (unsigned*)dst);
  };
  cvt(x, xbf, (long)kM * kH);
  cvt(l0_Wih, w_l0_Wih, 4 * (long)WSZ);
  cvt(l0_Whh, w_l0_Whh, 4 * (long)WSZ);
  cvt(fo_W, w_fo, (long)WSZ);
  cvt(fi_W, w_fi, (long)WSZ);
  cvt(l1_Wih, w_l1_Wih, (long)WSZ);
  cvt(l1_Whh, w_l1_Whh, (long)WSZ);

  auto gemm_blocks = [](int N) { return (kM / 16) * (N / 64) / 8; };

  auto run_lstm = [&](const __bf16* inseq, const __bf16* Wih,
                      const __bf16* Whh, const float* b, __bf16* outseq) {
    gemm_bias_act<kH, 0, false><<<gemm_blocks(4 * kH), 256, 0, stream>>>(
        inseq, Wih, b, xg, 4 * kH);
    (void)hipMemsetAsync(hbuf, 0, HBUF_BYTES, stream);
    (void)hipMemsetAsync(ctr, 0, sizeof(unsigned), stream);
    lstm_recurrence<<<16, 128, 0, stream>>>(xg, Whh, hbuf, outseq, ctr);
  };

  // 4-layer stacked LSTM: xbf -> seqA -> seqB -> seqA -> seqB
  const __bf16* cur = xbf;
  for (int l = 0; l < 4; ++l) {
    __bf16* dst = (l & 1) ? seqB : seqA;
    run_lstm(cur, w_l0_Wih + (size_t)l * WSZ, w_l0_Whh + (size_t)l * WSZ,
             l0_b + (size_t)l * (4 * kH), dst);
    cur = dst;
  }

  // fan_out (H->F) + LeakyReLU into bf16 mid (aliases xg region)
  gemm_bias_act<kH, 1, true><<<gemm_blocks(kF), 256, 0, stream>>>(
      cur, w_fo, fo_b, mid, kF);
  // fan_in (F->H) into bf16 seqA
  gemm_bias_act<kF, 0, true><<<gemm_blocks(kH), 256, 0, stream>>>(
      mid, w_fi, fi_b, seqA, kH);

  // final single-layer LSTM: seqA -> seqB
  run_lstm(seqA, w_l1_Wih, w_l1_Whh, l1_b, seqB);

  // head on last timestep
  head_kernel<<<1, 256, 0, stream>>>(seqB, hd_W, hd_b, out);
}
